// MHA_31636729102605
// MI455X (gfx1250) — compile-verified, matrix operand order checked by reading, not run on hardware
//
#include <hip/hip_runtime.h>
#include <hip/hip_bf16.h>

// MHA layer: B=4, S=2048, D=512, H=8, d_head(QK=V)=128.
// bf16 WMMA (v_wmma_f32_16x16x32_bf16) for all matmuls; flash-attention
// structure for score/softmax/PV. Weights pre-transposed and V produced
// in [b,h,v,S] layout so every fragment load vectorizes to b128.

#define BATCH  4
#define SEQ    2048
#define NH     8
#define DHEAD  128
#define DMODEL 512

typedef __attribute__((ext_vector_type(16))) __bf16 v16bf;
typedef __attribute__((ext_vector_type(8)))  float  v8f;

union FragB16 { v16bf v; unsigned int u[8]; };

__device__ __forceinline__ unsigned short f32_bf16(float f) {
  unsigned int u = __float_as_uint(f);
  u += 0x7FFFu + ((u >> 16) & 1u);   // round-to-nearest-even
  return (unsigned short)(u >> 16);
}

// ---------------------------------------------------------------------------
// fp32 -> bf16 conversion (grid-stride)
// ---------------------------------------------------------------------------
__global__ void cvt_f32_bf16(const float* __restrict__ src,
                             unsigned short* __restrict__ dst, int n) {
  int i = blockIdx.x * blockDim.x + threadIdx.x;
  int stride = gridDim.x * blockDim.x;
  for (; i < n; i += stride) dst[i] = f32_bf16(src[i]);
}

// fp32 [K,N] -> bf16 transposed [N,K] (so GEMM B-fragments load contiguously)
__global__ void cvt_t_f32_bf16(const float* __restrict__ src,
                               unsigned short* __restrict__ dst, int K, int N) {
  int i = blockIdx.x * blockDim.x + threadIdx.x;
  int stride = gridDim.x * blockDim.x;
  int total = K * N;
  for (; i < total; i += stride) {
    int k = i / N, n = i - k * N;
    dst[(size_t)n * K + k] = f32_bf16(src[i]);
  }
}

// ---------------------------------------------------------------------------
// bf16 GEMM: C[M,N] = A[M,K] * Bt[N,K]^T + bias[N]
// Wave tile 16(M) x 64(N): one A fragment reused across 4 WMMAs.
// 4 waves per 128-thread block -> block tile 16 x 256.
// STORE_MODE: 0 = f32 row-major, 1 = bf16 row-major, 2 = bf16 V-transposed
//             ([b,h,d,S]; M interpreted as b*SEQ+s, N as h*DHEAD+d)
// ---------------------------------------------------------------------------
template <int STORE_MODE>
__global__ void gemm_bf16_wmma(const unsigned short* __restrict__ A,
                               const unsigned short* __restrict__ Bt,
                               const float* __restrict__ bias,
                               void* __restrict__ Cout,
                               int M, int N, int K) {
  const int lane  = threadIdx.x & 31;
  const int wave  = threadIdx.x >> 5;
  const int lh    = lane >> 4;        // lane half (0/1)
  const int ll    = lane & 15;
  const int nbase = (blockIdx.x * 4 + wave) * 64;
  const int mbase = blockIdx.y * 16;
  const int arow  = mbase + ll;       // A-fragment: row = lane&15

  v8f acc[4];
#pragma unroll
  for (int t = 0; t < 4; ++t) acc[t] = (v8f){};

  for (int k0 = 0; k0 < K; k0 += 32) {
    FragB16 a;
#pragma unroll
    for (int v = 0; v < 8; ++v) {
      int kk = k0 + (v >> 2) * 16 + lh * 8 + (v & 3) * 2;
      a.u[v] = *(const unsigned int*)(A + (size_t)arow * K + kk);
    }
#pragma unroll
    for (int t = 0; t < 4; ++t) {
      const int bcol = nbase + t * 16 + ll;   // B-fragment: col = lane&15
      FragB16 b;
#pragma unroll
      for (int v = 0; v < 8; ++v) {
        int kk = k0 + (v >> 2) * 16 + lh * 8 + (v & 3) * 2;
        b.u[v] = *(const unsigned int*)(Bt + (size_t)bcol * K + kk);
      }
      acc[t] = __builtin_amdgcn_wmma_f32_16x16x32_bf16(false, a.v, false, b.v,
                                                       (short)0, acc[t], false, false);
    }
  }

#pragma unroll
  for (int t = 0; t < 4; ++t) {
    const int bcol = nbase + t * 16 + ll;
    const float bv = bias ? bias[bcol] : 0.0f;
#pragma unroll
    for (int r = 0; r < 8; ++r) {
      const int row = mbase + r + lh * 8;     // C/D layout: M = r + 8*lane_half
      const float val = acc[t][r] + bv;
      if (STORE_MODE == 0) {
        ((float*)Cout)[(size_t)row * N + bcol] = val;
      } else if (STORE_MODE == 1) {
        ((unsigned short*)Cout)[(size_t)row * N + bcol] = f32_bf16(val);
      } else {
        // V-transposed: [b, h, d, S]
        const int b_  = row >> 11;            // row / SEQ
        const int s_  = row & (SEQ - 1);
        const int h_  = bcol >> 7;            // col / DHEAD
        const int d_  = bcol & (DHEAD - 1);
        ((unsigned short*)Cout)[(((size_t)b_ * NH + h_) * DHEAD + d_) * SEQ + s_] =
            f32_bf16(val);
      }
    }
  }
}

// ---------------------------------------------------------------------------
// Flash attention: one wave per (batch, head, 16-query tile).
// Key tiles of 32; online softmax; scores never touch HBM.
//   S  = Q Kt / sqrt(128) + pos[h] + causal
//   O  = softmax(S) V     (bf16 out, [b,s,h,v] layout)
// Q,K in [b,s,h,k]; V in [b,h,v,S] so PV B-fragments are b128 loads.
// ---------------------------------------------------------------------------
__global__ void __launch_bounds__(32)
flash_attn_wmma(const unsigned short* __restrict__ Q,
                const unsigned short* __restrict__ Kt,
                const unsigned short* __restrict__ Vt,
                const float* __restrict__ pos,
                unsigned short* __restrict__ O) {
  __shared__ unsigned short p_lds[16 * 32];   // P tile bounce buffer (bf16)

  const int lane = threadIdx.x & 31;
  const int lh   = lane >> 4;
  const int ll   = lane & 15;

  const int bid = blockIdx.x;
  const int qt  = bid & (SEQ / 16 - 1);
  const int h   = (bid / (SEQ / 16)) & (NH - 1);
  const int b   = bid / ((SEQ / 16) * NH);
  const int qbase = qt * 16;

  // Load Q fragments: 16 x 128 bf16 = 4 A-fragments (K-chunks of 32)
  FragB16 qf[4];
  {
    const int s = qbase + ll;
    const unsigned short* qrow = Q + ((size_t)(b * SEQ + s) * NH + h) * DHEAD;
#pragma unroll
    for (int kc = 0; kc < 4; ++kc)
#pragma unroll
      for (int v = 0; v < 8; ++v) {
        int kk = kc * 32 + (v >> 2) * 16 + lh * 8 + (v & 3) * 2;
        qf[kc].u[v] = *(const unsigned int*)(qrow + kk);
      }
  }

  float mrow[8], lrow[8];
  v8f oacc[8];
#pragma unroll
  for (int r = 0; r < 8; ++r) { mrow[r] = -3.0e38f; lrow[r] = 0.0f; }
#pragma unroll
  for (int nc = 0; nc < 8; ++nc) oacc[nc] = (v8f){};

  const float scale = 0.08838834764831845f;   // 1/sqrt(128)

  // Causal: only key tiles touching or below the diagonal.
  for (int kb = 0; kb < qbase + 16; kb += 32) {
    // Prefetch next key tile (K rows + V columns) while we compute this one.
    if (kb + 32 < qbase + 16) {
      __builtin_prefetch(Kt + ((size_t)(b * SEQ + kb + 32 + ll) * NH + h) * DHEAD, 0, 1);
      __builtin_prefetch(Kt + ((size_t)(b * SEQ + kb + 48 + ll) * NH + h) * DHEAD, 0, 1);
      __builtin_prefetch(Vt + (((size_t)b * NH + h) * DHEAD + lane * 4) * SEQ + kb + 32, 0, 1);
    }

    // ---- S tile = Q * K^T (16 x 32) as two 16x16 column blocks ----
    v8f sacc[2]; sacc[0] = (v8f){}; sacc[1] = (v8f){};
#pragma unroll
    for (int c = 0; c < 2; ++c) {
      const int key = kb + c * 16 + ll;       // B-fragment: col = lane&15
      const unsigned short* krow = Kt + ((size_t)(b * SEQ + key) * NH + h) * DHEAD;
#pragma unroll
      for (int kc = 0; kc < 4; ++kc) {
        FragB16 kf;
#pragma unroll
        for (int v = 0; v < 8; ++v) {
          int kk = kc * 32 + (v >> 2) * 16 + lh * 8 + (v & 3) * 2;
          kf.u[v] = *(const unsigned int*)(krow + kk);  // contiguous K-dim pair
        }
        sacc[c] = __builtin_amdgcn_wmma_f32_16x16x32_bf16(
            false, qf[kc].v, false, kf.v, (short)0, sacc[c], false, false);
      }
    }

    // ---- scale + positional bias + causal mask + online softmax ----
    float p0[8], p1[8];
#pragma unroll
    for (int r = 0; r < 8; ++r) {
      const int sq  = qbase + r + lh * 8;     // C-layout row
      const int sk0 = kb + ll;
      const int sk1 = kb + 16 + ll;
      float v0 = sacc[0][r] * scale + pos[((size_t)h * SEQ + sq) * SEQ + sk0];
      float v1 = sacc[1][r] * scale + pos[((size_t)h * SEQ + sq) * SEQ + sk1];
      if (sk0 > sq) v0 = -1.0e9f;
      if (sk1 > sq) v1 = -1.0e9f;

      float tm = fmaxf(v0, v1);
#pragma unroll
      for (int off = 1; off < 16; off <<= 1)  // reduce within 16-lane row group
        tm = fmaxf(tm, __shfl_xor(tm, off, 32));
      const float mnew  = fmaxf(mrow[r], tm);
      const float alpha = __expf(mrow[r] - mnew);
      const float e0 = __expf(v0 - mnew);
      const float e1 = __expf(v1 - mnew);
      float rs = e0 + e1;
#pragma unroll
      for (int off = 1; off < 16; off <<= 1)
        rs += __shfl_xor(rs, off, 32);
      lrow[r] = lrow[r] * alpha + rs;
      mrow[r] = mnew;
      p0[r] = e0; p1[r] = e1;
#pragma unroll
      for (int nc = 0; nc < 8; ++nc) oacc[nc][r] *= alpha;
    }

    // ---- P (C-layout f32) -> LDS bf16 -> A-fragment layout ----
    __syncthreads();
#pragma unroll
    for (int r = 0; r < 8; ++r) {
      const int row = r + lh * 8;
      p_lds[row * 32 + ll]      = f32_bf16(p0[r]);
      p_lds[row * 32 + 16 + ll] = f32_bf16(p1[r]);
    }
    __syncthreads();
    FragB16 pf;
#pragma unroll
    for (int v = 0; v < 8; ++v) {
      int kk = (v >> 2) * 16 + lh * 8 + (v & 3) * 2;
      pf.u[v] = *(const unsigned int*)(&p_lds[ll * 32 + kk]);
    }

    // ---- O += P * V  (K = 32 keys, 8 N-chunks of 16 over d_head=128) ----
#pragma unroll
    for (int nc = 0; nc < 8; ++nc) {
      const unsigned short* vcol =
          Vt + (((size_t)b * NH + h) * DHEAD + nc * 16 + ll) * SEQ;  // keys contiguous
      FragB16 vf;
#pragma unroll
      for (int v = 0; v < 8; ++v) {
        int kk = (v >> 2) * 16 + lh * 8 + (v & 3) * 2;
        vf.u[v] = *(const unsigned int*)(vcol + kb + kk);
      }
      oacc[nc] = __builtin_amdgcn_wmma_f32_16x16x32_bf16(
          false, pf.v, false, vf.v, (short)0, oacc[nc], false, false);
    }
    __syncthreads();
  }

  // ---- normalize and store O (bf16, [b,s,h,v]) ----
#pragma unroll
  for (int r = 0; r < 8; ++r) {
    const int s = qbase + r + lh * 8;
    const float inv = 1.0f / lrow[r];
    unsigned short* orow = O + ((size_t)(b * SEQ + s) * NH + h) * DHEAD;
#pragma unroll
    for (int nc = 0; nc < 8; ++nc)
      orow[nc * 16 + ll] = f32_bf16(oacc[nc][r] * inv);
  }
}

// ---------------------------------------------------------------------------
// Host launcher
// ---------------------------------------------------------------------------
extern "C" void kernel_launch(void* const* d_in, const int* in_sizes, int n_in,
                              void* d_out, int out_size, void* d_ws, size_t ws_size,
                              hipStream_t stream) {
  (void)in_sizes; (void)n_in; (void)out_size; (void)ws_size;

  const float* x     = (const float*)d_in[0];
  /* d_in[1] = causal mask: reproduced analytically */
  const float* pos   = (const float*)d_in[2];
  const float* wq_w  = (const float*)d_in[3];
  const float* wq_b  = (const float*)d_in[4];
  const float* wk_w  = (const float*)d_in[5];
  const float* wk_b  = (const float*)d_in[6];
  const float* wv_w  = (const float*)d_in[7];
  const float* wv_b  = (const float*)d_in[8];
  const float* wo_w  = (const float*)d_in[9];
  const float* wo_b  = (const float*)d_in[10];
  float* out = (float*)d_out;

  const int M  = BATCH * SEQ;        // 8192
  const int HD = NH * DHEAD;         // 1024

  char* ws = (char*)d_ws;
  size_t off = 0;
  auto take = [&](size_t bytes) -> void* {
    void* p = ws + off;
    off = (off + bytes + 255) & ~(size_t)255;
    return p;
  };
  unsigned short* xb  = (unsigned short*)take((size_t)M * DMODEL * 2);
  unsigned short* wqb = (unsigned short*)take((size_t)DMODEL * HD * 2);  // [HD, D]
  unsigned short* wkb = (unsigned short*)take((size_t)DMODEL * HD * 2);  // [HD, D]
  unsigned short* wvb = (unsigned short*)take((size_t)DMODEL * HD * 2);  // [HD, D]
  unsigned short* wob = (unsigned short*)take((size_t)HD * DMODEL * 2);  // [D, HD]
  unsigned short* Qb  = (unsigned short*)take((size_t)M * HD * 2);       // [b,s,h,k]
  unsigned short* Kb  = (unsigned short*)take((size_t)M * HD * 2);       // [b,s,h,k]
  unsigned short* Vb  = (unsigned short*)take((size_t)M * HD * 2);       // [b,h,v,S]
  unsigned short* Ob  = (unsigned short*)take((size_t)M * HD * 2);       // [b,s,h,v]

  // 1) fp32 -> bf16 (weights transposed to [N,K] for contiguous B-fragments)
  cvt_f32_bf16  <<<4096, 256, 0, stream>>>(x,    xb,  M * DMODEL);
  cvt_t_f32_bf16<<<1024, 256, 0, stream>>>(wq_w, wqb, DMODEL, HD);
  cvt_t_f32_bf16<<<1024, 256, 0, stream>>>(wk_w, wkb, DMODEL, HD);
  cvt_t_f32_bf16<<<1024, 256, 0, stream>>>(wv_w, wvb, DMODEL, HD);
  cvt_t_f32_bf16<<<1024, 256, 0, stream>>>(wo_w, wob, HD, DMODEL);

  // 2) QKV projections (bf16 out; V written transposed to [b,h,v,S])
  dim3 blk(128);
  dim3 grid_qkv(HD / 256, M / 16);           // (4, 512)
  gemm_bf16_wmma<1><<<grid_qkv, blk, 0, stream>>>(xb, wqb, wq_b, Qb, M, HD, DMODEL);
  gemm_bf16_wmma<1><<<grid_qkv, blk, 0, stream>>>(xb, wkb, wk_b, Kb, M, HD, DMODEL);
  gemm_bf16_wmma<2><<<grid_qkv, blk, 0, stream>>>(xb, wvb, wv_b, Vb, M, HD, DMODEL);

  // 3) flash attention: one wave per (b, h, 16-query tile)
  flash_attn_wmma<<<BATCH * NH * (SEQ / 16), 32, 0, stream>>>(Qb, Kb, Vb, pos, Ob);

  // 4) output projection (f32 out)
  dim3 grid_o(DMODEL / 256, M / 16);         // (2, 512)
  gemm_bf16_wmma<0><<<grid_o, blk, 0, stream>>>(Ob, wob, wo_b, out, M, DMODEL, HD);
}
